// GCN_80307298501189
// MI455X (gfx1250) — compile-verified, matrix operand order checked by reading, not run on hardware
//
#include <hip/hip_runtime.h>
#include <hip/hip_bf16.h>
#include <stdint.h>

#define D      128
#define NLAYER 3
#define LSLOPE 0.03f
#define BNEPS  1e-5f
#define APAD   136   // LDS row stride in ushorts (272B) to rotate banks

typedef __attribute__((ext_vector_type(16))) __bf16         v16bf;
typedef __attribute__((ext_vector_type(16))) unsigned short v16us;
typedef __attribute__((ext_vector_type(8)))  unsigned short v8us;
typedef __attribute__((ext_vector_type(8)))  float          v8f;
typedef __attribute__((ext_vector_type(4)))  float          v4f;

__device__ __forceinline__ unsigned short f2bf(float f) {
    unsigned int u = __float_as_uint(f);
    u += 0x7FFFu + ((u >> 16) & 1u);            // round-to-nearest-even bf16
    return (unsigned short)(u >> 16);
}

// ---- weight prep: Wc[l][k][n] f32 -> Wt[l][n][k] bf16 (col-major for WMMA B) ----
__global__ void gcn_prep_w(const float* __restrict__ Wc, unsigned short* __restrict__ Wt,
                           int total) {
    int i = blockIdx.x * 256 + threadIdx.x;
    if (i >= total) return;
    int l = i / (D * D);
    int r = i - l * D * D;
    int k = r / D, n = r - k * D;
    Wt[(size_t)l * D * D + (size_t)n * D + k] = f2bf(Wc[i]);
}

// ---- degree / norm ----
__global__ void gcn_deg_init(float* __restrict__ deg, int n) {
    int i = blockIdx.x * 256 + threadIdx.x;
    if (i < n) deg[i] = 1.0f;                   // self-loop
}
__global__ void gcn_deg_count(const int* __restrict__ col, float* __restrict__ deg, int E) {
    int e = blockIdx.x * 256 + threadIdx.x;
    if (e < E) atomicAdd(&deg[col[e]], 1.0f);
}
__global__ void gcn_deg_rsqrt(float* __restrict__ deg, int n) {
    int i = blockIdx.x * 256 + threadIdx.x;
    if (i < n) deg[i] = rsqrtf(deg[i]);         // deg >= 1 always
}

// ---- GEMM: out[N,D] = in[N,D] @ W  via v_wmma_f32_16x16x32_bf16 ----
// Block = 256 threads = 8 waves; block computes 16 rows x 128 cols.
// A tile (16x128) is converted f32->bf16 ONCE into LDS, shared by all 8 waves.
// Each wave owns one 16x16 output tile (n0 = wave*16), K-loop 4 x 32.
__global__ __launch_bounds__(256) void gcn_gemm_wmma(const float* __restrict__ in,
                                                     const unsigned short* __restrict__ Wt,
                                                     float* __restrict__ out) {
    __shared__ unsigned short As[16 * APAD];

    const int tid = threadIdx.x;
    const int m0  = blockIdx.x * 16;

    // cooperative stage: thread t converts 8 contiguous elements of A
    {
        const int r = tid >> 4;             // 0..15
        const int c = (tid & 15) * 8;       // 0..120
        const float* src = in + (size_t)(m0 + r) * D + c;
        v4f a0 = *(const v4f*)(src);
        v4f a1 = *(const v4f*)(src + 4);
        v8us p;
#pragma unroll
        for (int j = 0; j < 4; ++j) { p[j] = f2bf(a0[j]); p[j + 4] = f2bf(a1[j]); }
        *(v8us*)(As + r * APAD + c) = p;    // 16B ds_store
    }
    __syncthreads();

    const int wave = tid >> 5;
    const int lane = tid & 31;
    const int half = lane >> 4;             // 0: K{0..7,16..23}  1: K{8..15,24..31}
    const int l16  = lane & 15;
    const int n0   = wave * 16;

    const unsigned short* arow = As + l16 * APAD;
    const unsigned short* bcol = Wt + (size_t)(n0 + l16) * D;

    v8f c = {};
#pragma unroll
    for (int k0 = 0; k0 < D; k0 += 32) {
        const int ka = k0 + 8 * half;
        v8us alo = *(const v8us*)(arow + ka);        // ds_load_b128
        v8us ahi = *(const v8us*)(arow + ka + 16);
        v8us blo = *(const v8us*)(bcol + ka);        // global_load_b128
        v8us bhi = *(const v8us*)(bcol + ka + 16);
        v16us au, bu;
#pragma unroll
        for (int j = 0; j < 8; ++j) {
            au[j] = alo[j]; au[j + 8] = ahi[j];
            bu[j] = blo[j]; bu[j + 8] = bhi[j];
        }
        c = __builtin_amdgcn_wmma_f32_16x16x32_bf16(
                false, __builtin_bit_cast(v16bf, au),
                false, __builtin_bit_cast(v16bf, bu),
                (short)0, c, false, false);
    }
    // C/D layout: VGPR j -> row m0 + j + 8*half, col n0 + l16
    float* orow = out + (size_t)(m0 + 8 * half) * D + (n0 + l16);
#pragma unroll
    for (int j = 0; j < 8; ++j) orow[(size_t)j * D] = c[j];
}

// ---- self-loop init: acc[i][d] = t[i][d] * dis[i]^2 ----
__global__ void gcn_self_init(const float* __restrict__ t, const float* __restrict__ dis,
                              float* __restrict__ acc, int total) {
    int i = blockIdx.x * 256 + threadIdx.x;
    if (i >= total) return;
    int node = i >> 7;
    float s = dis[node];
    acc[i] = t[i] * (s * s);
}

// ---- edge scatter: one wave per edge, lane owns 4 floats (coalesced 512B row) ----
__global__ __launch_bounds__(256) void gcn_scatter(const float* __restrict__ t,
                                                   const float* __restrict__ dis,
                                                   const int* __restrict__ row,
                                                   const int* __restrict__ col,
                                                   float* __restrict__ acc, int E) {
    int e = blockIdx.x * 8 + (threadIdx.x >> 5);
    if (e >= E) return;
    int lane = threadIdx.x & 31;
    int r = row[e], cdst = col[e];
    float nrm = dis[r] * dis[cdst];
    v4f v = *(const v4f*)(t + (size_t)r * D + lane * 4);
    v *= nrm;
    float* dst = acc + (size_t)cdst * D + lane * 4;
    atomicAdd(dst + 0, v[0]);
    atomicAdd(dst + 1, v[1]);
    atomicAdd(dst + 2, v[2]);
    atomicAdd(dst + 3, v[3]);
}

// ---- bias + (optional BN) + LeakyReLU, in place ----
__global__ void gcn_finalize(float* __restrict__ h, const float* __restrict__ bias,
                             const float* __restrict__ gamma, const float* __restrict__ beta,
                             const float* __restrict__ mean, const float* __restrict__ var,
                             int use_bn, int total) {
    int i = blockIdx.x * 256 + threadIdx.x;
    if (i >= total) return;
    int d = i & (D - 1);
    float v = h[i] + bias[d];
    if (use_bn) v = (v - mean[d]) * rsqrtf(var[d] + BNEPS) * gamma[d] + beta[d];
    h[i] = v > 0.0f ? v : LSLOPE * v;
}

// ---- pooling ----
__global__ void gcn_pool_clear(float* __restrict__ psum, int* __restrict__ pcnt, int gd, int g) {
    int i = blockIdx.x * 256 + threadIdx.x;
    if (i < gd) psum[i] = 0.0f;
    if (i < g)  pcnt[i] = 0;
}
__global__ __launch_bounds__(256) void gcn_pool_accum(const float* __restrict__ h,
                                                      const int* __restrict__ batch,
                                                      float* __restrict__ psum,
                                                      int* __restrict__ pcnt, int N) {
    int i = blockIdx.x * 8 + (threadIdx.x >> 5);
    if (i >= N) return;
    int lane = threadIdx.x & 31;
    int g = batch[i];
    v4f v = *(const v4f*)(h + (size_t)i * D + lane * 4);
    float* dst = psum + (size_t)g * D + lane * 4;
    atomicAdd(dst + 0, v[0]);
    atomicAdd(dst + 1, v[1]);
    atomicAdd(dst + 2, v[2]);
    atomicAdd(dst + 3, v[3]);
    if (lane == 0) atomicAdd(&pcnt[g], 1);
}
__global__ void gcn_head(const float* __restrict__ psum, const int* __restrict__ pcnt,
                         const float* __restrict__ lw, const float* __restrict__ lb,
                         float* __restrict__ out, int G, int C) {
    int id = blockIdx.x * 256 + threadIdx.x;
    if (id >= G * C) return;
    int g = id / C, c = id - g * C;
    float inv = 1.0f / fmaxf((float)pcnt[g], 1.0f);
    const float* p = psum + (size_t)g * D;
    const float* w = lw + (size_t)c * D;
    float s = 0.0f;
#pragma unroll 4
    for (int d = 0; d < D; ++d) s += p[d] * w[d];
    out[id] = s * inv + lb[c];
}

static inline size_t alignup(size_t x) { return (x + 255) & ~(size_t)255; }

extern "C" void kernel_launch(void* const* d_in, const int* in_sizes, int n_in,
                              void* d_out, int out_size, void* d_ws, size_t ws_size,
                              hipStream_t stream) {
    const float* x     = (const float*)d_in[0];
    const int*   eidx  = (const int*)d_in[1];     // [2,E] row-major: row then col
    const int*   batch = (const int*)d_in[2];
    const float* Wc    = (const float*)d_in[3];
    const float* bc    = (const float*)d_in[4];
    const float* bn_g  = (const float*)d_in[5];
    const float* bn_b  = (const float*)d_in[6];
    const float* bn_m  = (const float*)d_in[7];
    const float* bn_v  = (const float*)d_in[8];
    const float* lin_w = (const float*)d_in[9];
    const float* lin_b = (const float*)d_in[10];
    float* out = (float*)d_out;

    const int N = in_sizes[0] / D;
    const int E = in_sizes[1] / 2;
    const int C = in_sizes[10];
    const int G = out_size / C;
    const int* row = eidx;
    const int* col = eidx + E;
    const int ND = N * D;

    // workspace carve-up
    char* ws = (char*)d_ws;
    float* dis = (float*)ws;                  ws += alignup((size_t)N * 4);
    unsigned short* Wt = (unsigned short*)ws; ws += alignup((size_t)NLAYER * D * D * 2);
    float* hbuf = (float*)ws;                 ws += alignup((size_t)ND * 4);
    float* tbuf = (float*)ws;                 ws += alignup((size_t)ND * 4);
    float* psum = (float*)ws;                 ws += alignup((size_t)G * D * 4);
    int*   pcnt = (int*)ws;
    (void)ws_size; (void)n_in;

    // weights -> bf16, transposed
    gcn_prep_w<<<(NLAYER * D * D + 255) / 256, 256, 0, stream>>>(Wc, Wt, NLAYER * D * D);
    // gcn_norm
    gcn_deg_init<<<(N + 255) / 256, 256, 0, stream>>>(dis, N);
    gcn_deg_count<<<(E + 255) / 256, 256, 0, stream>>>(col, dis, E);
    gcn_deg_rsqrt<<<(N + 255) / 256, 256, 0, stream>>>(dis, N);

    for (int l = 0; l < NLAYER; ++l) {
        const float* in_feat = (l == 0) ? x : hbuf;
        gcn_gemm_wmma<<<N / 16, 256, 0, stream>>>(in_feat, Wt + (size_t)l * D * D, tbuf);
        gcn_self_init<<<(ND + 255) / 256, 256, 0, stream>>>(tbuf, dis, hbuf, ND);
        gcn_scatter<<<(E + 7) / 8, 256, 0, stream>>>(tbuf, dis, row, col, hbuf, E);
        int use_bn = (l < 2) ? 1 : 0;
        int bi = use_bn ? l : 0;
        gcn_finalize<<<(ND + 255) / 256, 256, 0, stream>>>(
            hbuf, bc + (size_t)l * D,
            bn_g + (size_t)bi * D, bn_b + (size_t)bi * D,
            bn_m + (size_t)bi * D, bn_v + (size_t)bi * D, use_bn, ND);
    }

    gcn_pool_clear<<<(G * D + 255) / 256, 256, 0, stream>>>(psum, pcnt, G * D, G);
    gcn_pool_accum<<<(N + 7) / 8, 256, 0, stream>>>(hbuf, batch, psum, pcnt, N);
    gcn_head<<<(G * C + 255) / 256, 256, 0, stream>>>(psum, pcnt, lin_w, lin_b, out, G, C);
}
